// Decoder_46119358825195
// MI455X (gfx1250) — compile-verified, matrix operand order checked by reading, not run on hardware
//
#include <hip/hip_runtime.h>
#include <stdint.h>

#define PADV -11.5129251f
#define TSP 408   // padded time stride for postnet buffers (t stored at pos t+2)

typedef __attribute__((ext_vector_type(16))) __bf16 bf16x16;
typedef __attribute__((ext_vector_type(8)))  float  f32x8;

// ---------------------------------------------------------------------------
// WMMA 16x32 bf16 tile loader. Rows are M (A) or N (B, weights stored [n][k]).
// Lane layout (16-bit operands, wave32): lanes 0-15 row=lane, K {0..7,16..23};
// lanes 16-31 row=lane-16, K {8..15,24..31}. Two aligned 16B loads per lane.
// Requires: stride % 8 == 0, k0step % 32 == 0, base 16B-aligned.
// ---------------------------------------------------------------------------
__device__ __forceinline__ bf16x16 load_tile(const __bf16* base, int stride,
                                             int row0, int k0step) {
  int lane = threadIdx.x & 31;
  int r = row0 + (lane & 15);
  int k0 = k0step + ((lane >> 4) << 3);
  const __bf16* p = base + (size_t)r * stride + k0;
  union { bf16x16 v; uint4 q[2]; } u;
  u.q[0] = *reinterpret_cast<const uint4*>(p);
  u.q[1] = *reinterpret_cast<const uint4*>(p + 16);
  return u.v;
}

// ---------------------------------------------------------------------------
// Fused LSTM cell: gates = A[32xK] * W[4096xK]^T (+bias), then pointwise.
// grid.x = 16 (64 hidden units each), 256 threads = 8 waves.
// Wave w: mtile = w&1 (batch rows 0-15/16-31), 4 n-tiles of 16 gate columns.
// k-loop unrolled x2 with ping-pong tile sets: loads stay one k-step ahead
// of the WMMAs and no register copies are needed between iterations.
// Requires K % 64 == 0 (1792 and 2560 both qualify).
// ---------------------------------------------------------------------------
__global__ __launch_bounds__(256)
void k_lstm(const __bf16* __restrict__ A, const __bf16* __restrict__ W,
            const float* __restrict__ bias,
            float* __restrict__ h_f32, float* __restrict__ c_f32,
            __bf16* __restrict__ hbf1, int s1, int o1,
            __bf16* __restrict__ hbf2, int s2, int o2, int K) {
  __shared__ float gates[32][257];
  int h0   = blockIdx.x * 64;
  int tid  = threadIdx.x;
  int wave = tid >> 5;
  int mt   = wave & 1;
  int nq   = wave >> 1;
  int grow[4];
#pragma unroll
  for (int q = 0; q < 4; ++q) {
    int ln  = (nq * 4 + q) * 16;
    grow[q] = ((ln >> 6) << 10) + h0 + (ln & 63);
  }
  f32x8 acc[4] = {};
  bf16x16 a0, b0[4], a1, b1[4];
  a0 = load_tile(A, K, mt * 16, 0);
#pragma unroll
  for (int q = 0; q < 4; ++q) b0[q] = load_tile(W, K, grow[q], 0);
  int ks = 0;
  for (; ks + 64 < K; ks += 64) {
    a1 = load_tile(A, K, mt * 16, ks + 32);
#pragma unroll
    for (int q = 0; q < 4; ++q) b1[q] = load_tile(W, K, grow[q], ks + 32);
#pragma unroll
    for (int q = 0; q < 4; ++q)
      acc[q] = __builtin_amdgcn_wmma_f32_16x16x32_bf16(
          false, a0, false, b0[q], (short)0, acc[q], false, false);
    a0 = load_tile(A, K, mt * 16, ks + 64);
#pragma unroll
    for (int q = 0; q < 4; ++q) b0[q] = load_tile(W, K, grow[q], ks + 64);
#pragma unroll
    for (int q = 0; q < 4; ++q)
      acc[q] = __builtin_amdgcn_wmma_f32_16x16x32_bf16(
          false, a1, false, b1[q], (short)0, acc[q], false, false);
  }
  // tail: set0 holds ks, load set1 at ks+32, consume both
  a1 = load_tile(A, K, mt * 16, ks + 32);
#pragma unroll
  for (int q = 0; q < 4; ++q) b1[q] = load_tile(W, K, grow[q], ks + 32);
#pragma unroll
  for (int q = 0; q < 4; ++q)
    acc[q] = __builtin_amdgcn_wmma_f32_16x16x32_bf16(
        false, a0, false, b0[q], (short)0, acc[q], false, false);
#pragma unroll
  for (int q = 0; q < 4; ++q)
    acc[q] = __builtin_amdgcn_wmma_f32_16x16x32_bf16(
        false, a1, false, b1[q], (short)0, acc[q], false, false);

  int lane = tid & 31, half = lane >> 4, col = lane & 15;
#pragma unroll
  for (int q = 0; q < 4; ++q) {
    int ln0 = (nq * 4 + q) * 16;
#pragma unroll
    for (int r = 0; r < 8; ++r)
      gates[mt * 16 + half * 8 + r][ln0 + col] = acc[q][r];
  }
  __syncthreads();
  for (int idx = tid; idx < 32 * 64; idx += 256) {
    int b = idx >> 6, u = idx & 63, gu = h0 + u;
    float gi = gates[b][u]       + bias[gu];
    float gf = gates[b][64 + u]  + bias[1024 + gu];
    float gg = gates[b][128 + u] + bias[2048 + gu];
    float go = gates[b][192 + u] + bias[3072 + gu];
    float si = 1.f / (1.f + __expf(-gi));
    float sf = 1.f / (1.f + __expf(-gf));
    float so = 1.f / (1.f + __expf(-go));
    float c2 = sf * c_f32[b * 1024 + gu] + si * tanhf(gg);
    float hv = so * tanhf(c2);
    c_f32[b * 1024 + gu] = c2;
    h_f32[b * 1024 + gu] = hv;
    if (hbf1) hbf1[(size_t)b * s1 + o1 + gu] = (__bf16)hv;
    if (hbf2) hbf2[(size_t)b * s2 + o2 + gu] = (__bf16)hv;
  }
}

// ---------------------------------------------------------------------------
// Location-sensitive attention. One workgroup per batch row (grid = 32).
// thread s in [0,256): conv(k=31) over [aw; awc], proj to 128, energy,
// wave32 shuffle softmax, context (512 dims), ctx staging for both GEMMs.
// ---------------------------------------------------------------------------
__global__ __launch_bounds__(256)
void k_attn(const float* __restrict__ ah, const float* __restrict__ memory,
            const float* __restrict__ amem, const int* __restrict__ mlen,
            const float* __restrict__ Wq, const float* __restrict__ vvec,
            const float* __restrict__ lconv, const float* __restrict__ lproj,
            float* __restrict__ aw_g, float* __restrict__ awc_g,
            float* __restrict__ ctx_f32,
            __bf16* __restrict__ Aa, __bf16* __restrict__ Ad,
            float* __restrict__ out_attn, int t) {
  __shared__ float sA[286], sB[286], pq[128], locf[256][33], awn[256];
  __shared__ float hbuf[1024], red[8];
  int b = blockIdx.x, tid = threadIdx.x;
  if (tid < 15) { sA[tid] = 0; sB[tid] = 0; sA[271 + tid] = 0; sB[271 + tid] = 0; }
  sA[15 + tid] = aw_g[b * 256 + tid];
  sB[15 + tid] = awc_g[b * 256 + tid];
  for (int j = tid; j < 1024; j += 256) hbuf[j] = ah[b * 1024 + j];
  __syncthreads();
  if (tid < 128) {
    const float* wr = Wq + tid * 1024;
    float s = 0;
    for (int j = 0; j < 1024; ++j) s += wr[j] * hbuf[j];
    pq[tid] = s;
  }
  { // location conv: all 32 filters for this thread's s
    int s = tid;
    for (int f = 0; f < 32; ++f) {
      const float* w0 = lconv + f * 62;
      const float* w1 = w0 + 31;
      float acc = 0;
      for (int k = 0; k < 31; ++k) acc += w0[k] * sA[s + k] + w1[k] * sB[s + k];
      locf[s][f] = acc;
    }
  }
  __syncthreads();
  float e;
  { // energy
    int s = tid;
    const float* am = amem + ((size_t)b * 256 + s) * 128;
    float acc = 0;
    for (int d = 0; d < 128; ++d) {
      float x = pq[d] + am[d];
      const float* lp = lproj + d * 32;
      for (int f = 0; f < 32; ++f) x += lp[f] * locf[s][f];
      acc += vvec[d] * tanhf(x);
    }
    e = (s < mlen[b]) ? acc : -1e30f;
  }
  // softmax: wave32 butterfly + 8-wave LDS reduce
  float mx = e;
#pragma unroll
  for (int o = 16; o; o >>= 1) mx = fmaxf(mx, __shfl_xor(mx, o, 32));
  if ((tid & 31) == 0) red[tid >> 5] = mx;
  __syncthreads();
  mx = red[0];
#pragma unroll
  for (int i = 1; i < 8; ++i) mx = fmaxf(mx, red[i]);
  float p = __expf(e - mx);
  float sm = p;
#pragma unroll
  for (int o = 16; o; o >>= 1) sm += __shfl_xor(sm, o, 32);
  __syncthreads();
  if ((tid & 31) == 0) red[tid >> 5] = sm;
  __syncthreads();
  float sum = red[0];
#pragma unroll
  for (int i = 1; i < 8; ++i) sum += red[i];
  float a = p / sum;
  awn[tid] = a;
  aw_g[b * 256 + tid]  = a;
  awc_g[b * 256 + tid] = sB[15 + tid] + a;
  out_attn[(size_t)(b * 256 + tid) * 401 + t] = a;
  __syncthreads();
  for (int d = tid; d < 512; d += 256) {
    const float* mp = memory + (size_t)b * 256 * 512 + d;
    float acc = 0;
    for (int s = 0; s < 256; ++s) acc += awn[s] * mp[s * 512];
    ctx_f32[b * 512 + d] = acc;
    Aa[(size_t)b * 1792 + 256 + d]  = (__bf16)acc;
    Ad[(size_t)b * 2560 + 1024 + d] = (__bf16)acc;
  }
}

// mel + stop projections: pred = [dh(1024), ctx(512)] -> 80 mel + 1 stop
__global__ __launch_bounds__(128)
void k_melstop(const float* __restrict__ dh, const float* __restrict__ ctx,
               const float* __restrict__ Wmel, const float* __restrict__ bmel,
               const float* __restrict__ Wstop, const float* __restrict__ bstop,
               float* __restrict__ out, int t) {
  __shared__ float pred[1536];
  int b = blockIdx.x, tid = threadIdx.x;
  for (int k = tid; k < 1024; k += 128) pred[k] = dh[b * 1024 + k];
  for (int k = tid; k < 512; k += 128) pred[1024 + k] = ctx[b * 512 + k];
  __syncthreads();
  if (tid < 80) {
    const float* w = Wmel + tid * 1536;
    float s = bmel[tid];
    for (int k = 0; k < 1536; ++k) s += w[k] * pred[k];
    if (t < 400) out[((size_t)b * 400 + t) * 80 + tid] = s;
  } else if (tid == 80) {
    float s = bstop[0];
    for (int k = 0; k < 1536; ++k) s += Wstop[k] * pred[k];
    out[2048000 + b * 401 + t] = 1.f / (1.f + __expf(-s));
  }
}

// per-step staging: pre_t, ah_{t-1}, dh_{t-1} into bf16 GEMM A buffers
__global__ void k_stage_step(const __bf16* __restrict__ pre_bf,
                             const __bf16* __restrict__ ah_bf,
                             const __bf16* __restrict__ dh_bf,
                             __bf16* __restrict__ Aa, __bf16* __restrict__ Ad,
                             int t) {
  int b = blockIdx.x, tid = threadIdx.x;
  Aa[(size_t)b * 1792 + tid] = pre_bf[((size_t)t * 32 + b) * 256 + tid];
  for (int j = tid; j < 1024; j += 256) {
    Aa[(size_t)b * 1792 + 768 + j]  = ah_bf[b * 1024 + j];
    Ad[(size_t)b * 2560 + 1536 + j] = dh_bf[b * 1024 + j];
  }
}

// prenet for all steps (one block per (t,b))
__global__ __launch_bounds__(256)
void k_prenet(const float* __restrict__ din, const float* __restrict__ W1,
              const float* __restrict__ b1, const float* __restrict__ W2,
              const float* __restrict__ b2, __bf16* __restrict__ pre_bf) {
  __shared__ float x[80], h1[256];
  int bid = blockIdx.x, t = bid >> 5, b = bid & 31, tid = threadIdx.x;
  if (tid < 80)
    x[tid] = (t == 0) ? PADV : din[((size_t)b * 400 + (t - 1)) * 80 + tid];
  __syncthreads();
  float s = b1[tid];
  const float* w = W1 + tid * 80;
  for (int k = 0; k < 80; ++k) s += w[k] * x[k];
  h1[tid] = fmaxf(s, 0.f);
  __syncthreads();
  float s2 = b2[tid];
  const float* w2 = W2 + tid * 256;
  for (int k = 0; k < 256; ++k) s2 += w2[k] * h1[k];
  pre_bf[((size_t)t * 32 + b) * 256 + tid] = (__bf16)fmaxf(s2, 0.f);
}

// attn_memory = memory @ Wm^T  (one-time)
__global__ __launch_bounds__(128)
void k_amem(const float* __restrict__ memory, const float* __restrict__ Wm,
            float* __restrict__ amem) {
  int bs = blockIdx.x, d = threadIdx.x;
  const float* m = memory + (size_t)bs * 512;
  const float* w = Wm + d * 512;
  float s = 0;
  for (int k = 0; k < 512; ++k) s += w[k] * m[k];
  amem[(size_t)bs * 128 + d] = s;
}

// weight prep
__global__ void k_cvt_lstm_w(const float* __restrict__ Wih,
                             const float* __restrict__ Whh,
                             __bf16* __restrict__ dst, int Kih, int Khh) {
  int n = blockIdx.x, K = Kih + Khh;
  for (int k = threadIdx.x; k < K; k += blockDim.x) {
    float v = (k < Kih) ? Wih[(size_t)n * Kih + k]
                        : Whh[(size_t)n * Khh + (k - Kih)];
    dst[(size_t)n * K + k] = (__bf16)v;
  }
}
__global__ void k_bias_sum(const float* a, const float* b, float* o, int n) {
  int i = blockIdx.x * 256 + threadIdx.x;
  if (i < n) o[i] = a[i] + b[i];
}
__global__ void k_cvt_post_w(const float* __restrict__ src,
                             __bf16* __restrict__ dst, int CO, int K,
                             int Kstride) {
  int co = blockIdx.x;
  for (int k = threadIdx.x; k < Kstride; k += blockDim.x) {
    float v = (co < CO && k < K) ? src[(size_t)co * K + k] : 0.f;
    dst[(size_t)co * Kstride + k] = (__bf16)v;
  }
}
__global__ void k_post_in(const float* __restrict__ mel,
                          __bf16* __restrict__ xb) {
  int b = blockIdx.x, m = blockIdx.y;
  for (int pos = threadIdx.x; pos < TSP; pos += blockDim.x) {
    float v = (pos >= 2 && pos < 402)
                  ? mel[((size_t)b * 400 + (pos - 2)) * 80 + m] : 0.f;
    xb[((size_t)b * 80 + m) * TSP + pos] = (__bf16)v;
  }
}

// ---------------------------------------------------------------------------
// Postnet conv layer as im2col GEMM: D[co][t] = W[co][ci*5+k] * x[ci][t+k].
// grid = (tchunks, co_chunks, B); each WG: 64 co x 64 t, K staged in LDS
// in ci-blocks of 128. Weight b128 loads issued BEFORE the LDS im2col gather
// so global traffic overlaps the ds_load assembly. bf16 WMMA + fused BN.
// ---------------------------------------------------------------------------
__global__ __launch_bounds__(256)
void k_post_layer(const __bf16* __restrict__ xb, const __bf16* __restrict__ wbf,
                  const float* __restrict__ bias, const float* __restrict__ gam,
                  const float* __restrict__ bet, __bf16* __restrict__ yb,
                  float* __restrict__ out, int CI, int CO, int Kstride,
                  int last) {
  __shared__ __bf16 xs[128][68];
  int t0 = blockIdx.x * 64, co0 = blockIdx.y * 64, b = blockIdx.z;
  int tid = threadIdx.x, wave = tid >> 5, lane = tid & 31;
  int ttile = wave & 3, cbase = (wave >> 2) * 2;
  f32x8 acc[2] = {};
  int nblk = (CI + 127) / 128;
  for (int cb = 0; cb < nblk; ++cb) {
    int ciN = CI - cb * 128;
    if (ciN > 128) ciN = 128;
    __syncthreads();
    for (int idx = tid; idx < 128 * 68; idx += 256) {
      int ci = idx / 68, j = idx % 68, pos = t0 + j;
      __bf16 v = (__bf16)0.f;
      if (ci < ciN && pos < TSP)
        v = xb[((size_t)b * CI + cb * 128 + ci) * TSP + pos];
      xs[ci][j] = v;
    }
    __syncthreads();
    int Kb = ciN * 5, nks = (Kb + 31) / 32;
    for (int ks = 0; ks < nks; ++ks) {
      // issue weight tile loads first (overlap with LDS gather below)
      bf16x16 a[2];
#pragma unroll
      for (int q = 0; q < 2; ++q)
        a[q] = load_tile(wbf, Kstride, co0 + (cbase + q) * 16,
                         cb * 640 + ks * 32);
      int tloc = ttile * 16 + (lane & 15);
      int half = lane >> 4;
      union { bf16x16 v; __bf16 e[16]; } bt;
#pragma unroll
      for (int j = 0; j < 16; ++j) {
        int vv = j >> 1;
        int kk = ks * 32 + ((vv < 4) ? 2 * vv : 16 + 2 * (vv - 4)) +
                 8 * half + (j & 1);
        __bf16 val = (__bf16)0.f;
        if (kk < Kb) val = xs[kk / 5][tloc + kk % 5];
        bt.e[j] = val;
      }
#pragma unroll
      for (int q = 0; q < 2; ++q)
        acc[q] = __builtin_amdgcn_wmma_f32_16x16x32_bf16(
            false, a[q], false, bt.v, (short)0, acc[q], false, false);
    }
  }
  const float inv = 0.9999950000374997f; // 1/sqrt(1+1e-5)
  int half = lane >> 4;
  int tabs = t0 + ttile * 16 + (lane & 15);
#pragma unroll
  for (int q = 0; q < 2; ++q) {
#pragma unroll
    for (int r = 0; r < 8; ++r) {
      int co = co0 + (cbase + q) * 16 + half * 8 + r;
      if (co >= CO || tabs >= 400) continue;
      float y = acc[q][r] + bias[co];
      y = gam[co] * y * inv + bet[co];
      if (!last) {
        yb[((size_t)b * CO + co) * TSP + (tabs + 2)] = (__bf16)tanhf(y);
      } else {
        size_t mi = ((size_t)b * 400 + tabs) * 80 + co;
        out[1024000 + mi] = out[mi] + y;
      }
    }
  }
}

// ---------------------------------------------------------------------------
extern "C" void kernel_launch(void* const* d_in, const int* in_sizes, int n_in,
                              void* d_out, int out_size, void* d_ws,
                              size_t ws_size, hipStream_t stream) {
  const float* memory = (const float*)d_in[0];
  const int*   mlen   = (const int*)d_in[1];
  const float* din    = (const float*)d_in[2];
  const float* pre_W1 = (const float*)d_in[4];
  const float* pre_b1 = (const float*)d_in[5];
  const float* pre_W2 = (const float*)d_in[6];
  const float* pre_b2 = (const float*)d_in[7];
  const float* aWih   = (const float*)d_in[8];
  const float* aWhh   = (const float*)d_in[9];
  const float* abih   = (const float*)d_in[10];
  const float* abhh   = (const float*)d_in[11];
  const float* dWih   = (const float*)d_in[12];
  const float* dWhh   = (const float*)d_in[13];
  const float* dbih   = (const float*)d_in[14];
  const float* dbhh   = (const float*)d_in[15];
  const float* Wq     = (const float*)d_in[16];
  const float* vvec   = (const float*)d_in[17];
  const float* lconv  = (const float*)d_in[18];
  const float* lproj  = (const float*)d_in[19];
  const float* Wm     = (const float*)d_in[20];
  const float* Wmel   = (const float*)d_in[21];
  const float* bmel   = (const float*)d_in[22];
  const float* Wstop  = (const float*)d_in[23];
  const float* bstop  = (const float*)d_in[24];
  const float* pconv[5]; const float* pb[5]; const float* pg[5]; const float* pbe[5];
  for (int i = 0; i < 5; ++i) {
    pconv[i] = (const float*)d_in[25 + 4 * i];
    pb[i]    = (const float*)d_in[26 + 4 * i];
    pg[i]    = (const float*)d_in[27 + 4 * i];
    pbe[i]   = (const float*)d_in[28 + 4 * i];
  }
  float* out = (float*)d_out;

  char* ws = (char*)d_ws;
  size_t off = 0;
  auto alloc = [&](size_t bytes) -> void* {
    off = (off + 255) & ~(size_t)255;
    void* p = ws + off;
    off += bytes;
    return p;
  };
  __bf16* wa     = (__bf16*)alloc((size_t)4096 * 1792 * 2);
  __bf16* wd     = (__bf16*)alloc((size_t)4096 * 2560 * 2);
  float*  ba     = (float*)alloc(4096 * 4);
  float*  bd     = (float*)alloc(4096 * 4);
  __bf16* pre_bf = (__bf16*)alloc((size_t)401 * 32 * 256 * 2);
  float*  amem   = (float*)alloc((size_t)32 * 256 * 128 * 4);
  __bf16* Aa     = (__bf16*)alloc((size_t)32 * 1792 * 2);
  __bf16* Ad     = (__bf16*)alloc((size_t)32 * 2560 * 2);
  float*  ahb    = (float*)alloc(32 * 1024 * 4);
  float*  acb    = (float*)alloc(32 * 1024 * 4);
  float*  dhb    = (float*)alloc(32 * 1024 * 4);
  float*  dcb    = (float*)alloc(32 * 1024 * 4);
  float*  ctxb   = (float*)alloc(32 * 512 * 4);
  float*  awb    = (float*)alloc(32 * 256 * 4);
  float*  awcb   = (float*)alloc(32 * 256 * 4);
  __bf16* ah_bf  = (__bf16*)alloc(32 * 1024 * 2);
  __bf16* dh_bf  = (__bf16*)alloc(32 * 1024 * 2);
  __bf16* wp0    = (__bf16*)alloc((size_t)512 * 416 * 2);
  __bf16* wp1    = (__bf16*)alloc((size_t)512 * 2560 * 2);
  __bf16* wp2    = (__bf16*)alloc((size_t)512 * 2560 * 2);
  __bf16* wp3    = (__bf16*)alloc((size_t)512 * 2560 * 2);
  __bf16* wp4    = (__bf16*)alloc((size_t)128 * 2560 * 2);
  __bf16* xb0    = (__bf16*)alloc((size_t)32 * 80 * TSP * 2);
  __bf16* xb1    = (__bf16*)alloc((size_t)32 * 512 * TSP * 2);
  __bf16* xb2    = (__bf16*)alloc((size_t)32 * 512 * TSP * 2);

  // zero-init state / staging / padded postnet buffers (graph-capturable)
  hipMemsetAsync(Aa,    0, (size_t)32 * 1792 * 2, stream);
  hipMemsetAsync(Ad,    0, (size_t)32 * 2560 * 2, stream);
  hipMemsetAsync(acb,   0, 32 * 1024 * 4, stream);
  hipMemsetAsync(dcb,   0, 32 * 1024 * 4, stream);
  hipMemsetAsync(awb,   0, 32 * 256 * 4, stream);
  hipMemsetAsync(awcb,  0, 32 * 256 * 4, stream);
  hipMemsetAsync(ah_bf, 0, 32 * 1024 * 2, stream);
  hipMemsetAsync(dh_bf, 0, 32 * 1024 * 2, stream);
  hipMemsetAsync(xb1,   0, (size_t)32 * 512 * TSP * 2, stream);
  hipMemsetAsync(xb2,   0, (size_t)32 * 512 * TSP * 2, stream);

  // one-time prep
  k_cvt_lstm_w<<<4096, 256, 0, stream>>>(aWih, aWhh, wa, 768, 1024);
  k_cvt_lstm_w<<<4096, 256, 0, stream>>>(dWih, dWhh, wd, 1536, 1024);
  k_bias_sum<<<16, 256, 0, stream>>>(abih, abhh, ba, 4096);
  k_bias_sum<<<16, 256, 0, stream>>>(dbih, dbhh, bd, 4096);
  k_prenet<<<401 * 32, 256, 0, stream>>>(din, pre_W1, pre_b1, pre_W2, pre_b2,
                                         pre_bf);
  k_amem<<<32 * 256, 128, 0, stream>>>(memory, Wm, amem);
  k_cvt_post_w<<<512, 256, 0, stream>>>(pconv[0], wp0, 512, 400, 416);
  k_cvt_post_w<<<512, 256, 0, stream>>>(pconv[1], wp1, 512, 2560, 2560);
  k_cvt_post_w<<<512, 256, 0, stream>>>(pconv[2], wp2, 512, 2560, 2560);
  k_cvt_post_w<<<512, 256, 0, stream>>>(pconv[3], wp3, 512, 2560, 2560);
  k_cvt_post_w<<<128, 256, 0, stream>>>(pconv[4], wp4, 80, 2560, 2560);

  // sequential decode: 401 steps
  for (int t = 0; t < 401; ++t) {
    k_stage_step<<<32, 256, 0, stream>>>(pre_bf, ah_bf, dh_bf, Aa, Ad, t);
    k_lstm<<<16, 256, 0, stream>>>(Aa, wa, ba, ahb, acb,
                                   Ad, 2560, 0, ah_bf, 1024, 0, 1792);
    k_attn<<<32, 256, 0, stream>>>(ahb, memory, amem, mlen, Wq, vvec, lconv,
                                   lproj, awb, awcb, ctxb, Aa, Ad,
                                   out + 2060832, t);
    k_lstm<<<16, 256, 0, stream>>>(Ad, wd, bd, dhb, dcb,
                                   dh_bf, 1024, 0, (__bf16*)nullptr, 0, 0,
                                   2560);
    k_melstop<<<32, 128, 0, stream>>>(dhb, ctxb, Wmel, bmel, Wstop, bstop,
                                      out, t);
  }

  // postnet (5 conv layers, WMMA GEMM)
  k_post_in<<<dim3(32, 80), 256, 0, stream>>>(out, xb0);
  k_post_layer<<<dim3(7, 8, 32), 256, 0, stream>>>(xb0, wp0, pb[0], pg[0],
      pbe[0], xb1, (float*)nullptr, 80, 512, 416, 0);
  k_post_layer<<<dim3(7, 8, 32), 256, 0, stream>>>(xb1, wp1, pb[1], pg[1],
      pbe[1], xb2, (float*)nullptr, 512, 512, 2560, 0);
  k_post_layer<<<dim3(7, 8, 32), 256, 0, stream>>>(xb2, wp2, pb[2], pg[2],
      pbe[2], xb1, (float*)nullptr, 512, 512, 2560, 0);
  k_post_layer<<<dim3(7, 8, 32), 256, 0, stream>>>(xb1, wp3, pb[3], pg[3],
      pbe[3], xb2, (float*)nullptr, 512, 512, 2560, 0);
  k_post_layer<<<dim3(7, 2, 32), 256, 0, stream>>>(xb2, wp4, pb[4], pg[4],
      pbe[4], (__bf16*)nullptr, out, 512, 80, 2560, 1);
}
